// bdnn2_27522150433389
// MI455X (gfx1250) — matrix operand order repaired and checked by reading, not compiled, not run on hardware
//
#include <hip/hip_runtime.h>

typedef __attribute__((ext_vector_type(16))) _Float16 v16h;
typedef __attribute__((ext_vector_type(8)))  _Float16 v8h;
typedef __attribute__((ext_vector_type(4)))  _Float16 v4h;
typedef __attribute__((ext_vector_type(8)))  float    v8f;
typedef __attribute__((ext_vector_type(4)))  unsigned int v4u;
typedef __attribute__((ext_vector_type(4)))  int      v4i;

// Problem dims (from reference setup_inputs)
constexpr int N_ = 2048;   // batch
constexpr int L_ = 100;    // timesteps
constexpr int D_ = 512;    // input features
constexpr int T_ = 1024;   // hidden / output
constexpr float ALPHA_ = 0.25f;
constexpr float INV_L_ = 1.0f / 100.0f;

constexpr int LDS_STRIDE = 40;   // padded row stride (f16 elems): 16 lanes -> 16 distinct banks

// ---- CDNA5 async global->LDS copy (ASYNCcnt-tracked), with safe fallback ----
#if defined(__HIP_DEVICE_COMPILE__) && __has_builtin(__builtin_amdgcn_global_load_async_to_lds_b128)
#define HAVE_ASYNC_LDS 1
#endif
#if defined(__HIP_DEVICE_COMPILE__) && __has_builtin(__builtin_amdgcn_s_wait_asynccnt)
#define HAVE_WAIT_ASYNC 1
#endif

__device__ __forceinline__ void cp16_g2l(_Float16* l, const _Float16* g) {
#if defined(HAVE_ASYNC_LDS)
    // global_load_async_to_lds_b128: (src v4i*, lds dst v4i*, imm offset, imm cpol)
    __builtin_amdgcn_global_load_async_to_lds_b128((v4i*)g, (v4i*)l, 0, 0);
#else
    *reinterpret_cast<v4u*>(l) = *reinterpret_cast<const v4u*>(g);   // vmem load + ds_store_b128
#endif
}

__device__ __forceinline__ void wait_async_copies() {
#if defined(HAVE_ASYNC_LDS)
#if defined(HAVE_WAIT_ASYNC)
    __builtin_amdgcn_s_wait_asynccnt(0);
#else
    asm volatile("s_wait_asynccnt 0" ::: "memory");
#endif
#endif
}

// ---- WMMA fragment helpers (16-bit A / B in contiguous k = 16*half + i order; f32 C layout) ----
__device__ __forceinline__ v16h read_a_lds(const _Float16* buf, int row, int half) {
    const _Float16* p = buf + row * LDS_STRIDE + half * 16;
    const v8h lo = *reinterpret_cast<const v8h*>(p);        // K = koff .. +7
    const v8h hi = *reinterpret_cast<const v8h*>(p + 8);    // K = koff+8 .. +15
    v16h a;
    #pragma unroll
    for (int i = 0; i < 8; ++i) { a[i] = lo[i]; a[i + 8] = hi[i]; }
    return a;
}

__device__ __forceinline__ v8f wmma_f16(v16h a, v16h b, v8f c) {
    return __builtin_amdgcn_wmma_f32_16x16x32_f16(false, a, false, b, (short)0, c, false, false);
}

// ---- double-buffered, async-staged K loop: accumulates a 32x32 per-wave tile ----
template <int NCH>
__device__ __forceinline__ void gemm_stage_loop(
    const _Float16* g_src,          // per-thread staging src (chunk 0): 16B segment
    _Float16* my_dst0, _Float16* my_dst1,   // per-thread staging dst in buf0 / buf1
    const _Float16* lds0, const _Float16* lds1,
    const _Float16* bp0, const _Float16* bp1,  // per-lane B fragment base (chunk 0)
    int wm, int lr, int half,
    v8f& c00, v8f& c01, v8f& c10, v8f& c11)
{
    cp16_g2l(my_dst0, g_src);
    wait_async_copies();
    __syncthreads();
    #pragma unroll 2
    for (int ch = 0; ch < NCH; ++ch) {
        const _Float16* cur = (ch & 1) ? lds1 : lds0;
        _Float16* nxt = (ch & 1) ? my_dst0 : my_dst1;
        if (ch + 1 < NCH) cp16_g2l(nxt, g_src + (ch + 1) * 32);   // overlap with compute
        __builtin_prefetch(bp0 + ch * 32 + 256, 0, 1);            // global_prefetch_b8
        const v16h a0 = read_a_lds(cur, wm + lr, half);           // ds_load_b128 x2
        const v16h a1 = read_a_lds(cur, wm + 16 + lr, half);
        const v16h b0 = *reinterpret_cast<const v16h*>(bp0 + (size_t)ch * 32);
        const v16h b1 = *reinterpret_cast<const v16h*>(bp1 + (size_t)ch * 32);
        c00 = wmma_f16(a0, b0, c00);
        c01 = wmma_f16(a0, b1, c01);
        c10 = wmma_f16(a1, b0, c10);
        c11 = wmma_f16(a1, b1, c11);
        if (ch + 1 < NCH) wait_async_copies();   // next chunk landed in LDS
        __syncthreads();                          // publish + release prev buffer
    }
}

// ---------------- f32 -> f16 convert (X), vectorized ----------------
__global__ __launch_bounds__(256) void cvt_x_kernel(const float* __restrict__ x,
                                                    _Float16* __restrict__ x16) {
    size_t i = ((size_t)blockIdx.x * 256 + threadIdx.x) * 4;
    const float4 v = *reinterpret_cast<const float4*>(x + i);
    v4h h;
    h[0] = (_Float16)v.x; h[1] = (_Float16)v.y;
    h[2] = (_Float16)v.z; h[3] = (_Float16)v.w;
    *reinterpret_cast<v4h*>(x16 + i) = h;
}

// ---------------- transpose + convert: src[rows][cols] f32 -> dst[cols][rows] f16 ----------------
__global__ __launch_bounds__(256) void transpose_cvt_kernel(const float* __restrict__ src,
                                                            _Float16* __restrict__ dst,
                                                            int rows, int cols) {
    int i = blockIdx.x * 256 + threadIdx.x;
    int r = i / cols;
    int c = i - r * cols;
    dst[(size_t)c * rows + r] = (_Float16)src[i];
}

// ---------------- zero workspace state ----------------
__global__ __launch_bounds__(256) void zero_kernel(v4u* __restrict__ p) {
    v4u z = {};
    p[(size_t)blockIdx.x * 256 + threadIdx.x] = z;
}

// ---------------- fused LIF step: h1 = X[:,t,:]@W + spk@U, then LIF update ----------------
// grid = (N/64, T/128), block = 256 = 8 waves arranged 2(M) x 4(N); 32x32 tile per wave.
__global__ __launch_bounds__(256)
void lif_step_kernel(const _Float16* __restrict__ X16,   // [N, L, D] f16
                     const _Float16* __restrict__ WT,    // [T, D] f16 (W^T)
                     const _Float16* __restrict__ UT,    // [T, T] f16 (U^T)
                     const _Float16* __restrict__ spk_in,// [N, T] f16 spikes of step t-1
                     _Float16* __restrict__ spk_out,     // [N, T] f16 spikes of step t
                     float* __restrict__ syn,
                     float* __restrict__ mem,
                     float* __restrict__ hsum,
                     int t) {
    __shared__ _Float16 lds_a[2][64 * LDS_STRIDE];   // double-buffered 64-row A chunk (10 KB)

    const int tid  = threadIdx.x;
    const int lane = tid & 31;
    const int wave = tid >> 5;
    const int bmG  = blockIdx.x * 64;                 // block row base (batch n)
    const int wm   = (wave >> 2) * 32;                // wave row offset in tile
    const int cbG  = blockIdx.y * 128 + (wave & 3) * 32;  // wave col base (hidden)

    const int lr   = lane & 15;
    const int half = lane >> 4;

    // staging decomposition: thread -> (row r of 64, 16-byte segment s of 4)
    const int sr = tid >> 2;
    const int ss = tid & 3;
    _Float16* my_dst0 = &lds_a[0][sr * LDS_STRIDE + ss * 8];
    _Float16* my_dst1 = &lds_a[1][sr * LDS_STRIDE + ss * 8];

    v8f c00 = {}, c01 = {}, c10 = {}, c11 = {};

    // ---- GEMM 1: X[:, t, :] @ W   (K = 512, 16 chunks) ----
    {
        const _Float16* g_src = X16 + ((size_t)(bmG + sr) * L_ + t) * D_ + ss * 8;
        const _Float16* bp0 = WT + (size_t)(cbG + lr) * D_ + half * 16;
        const _Float16* bp1 = WT + (size_t)(cbG + 16 + lr) * D_ + half * 16;
        gemm_stage_loop<D_ / 32>(g_src, my_dst0, my_dst1, lds_a[0], lds_a[1],
                                 bp0, bp1, wm, lr, half, c00, c01, c10, c11);
    }

    // ---- GEMM 2: spk_in @ U   (K = 1024, 32 chunks); A is exact {0,1} in f16 ----
    {
        const _Float16* g_src = spk_in + (size_t)(bmG + sr) * T_ + ss * 8;
        const _Float16* bp0 = UT + (size_t)(cbG + lr) * T_ + half * 16;
        const _Float16* bp1 = UT + (size_t)(cbG + 16 + lr) * T_ + half * 16;
        gemm_stage_loop<T_ / 32>(g_src, my_dst0, my_dst1, lds_a[0], lds_a[1],
                                 bp0, bp1, wm, lr, half, c00, c01, c10, c11);
    }

    // ---- LIF epilogue over the 4 sub-tiles. C layout: half -> M += 8, VGPR r -> row, N = lane&15.
    const int mb = half * 8;
    auto lif_tile = [&](const v8f& cc, int gm_base, int gn_base) {
        #pragma unroll
        for (int r = 0; r < 8; ++r) {
            const size_t idx = (size_t)(gm_base + mb + r) * T_ + (gn_base + lr);
            const float h1 = cc[r];                 // h_t + out_prev @ U
            const float s  = syn[idx];
            const float m  = mem[idx];
            const float out = ((m - 1.0f) > 0.0f) ? 1.0f : 0.0f;  // spike from PRE-update mem
            syn[idx] = ALPHA_ * s + h1;             // new_syn = alpha*syn + h1
            mem[idx] = (m + s) * (1.0f - out);      // new_mem = (mem + syn)*(1-rst), beta=1
            spk_out[idx] = (_Float16)out;           // exact in f16
            hsum[idx] += out;                       // unique owner per element
        }
    };
    lif_tile(c00, bmG + wm,      cbG);
    lif_tile(c01, bmG + wm,      cbG + 16);
    lif_tile(c10, bmG + wm + 16, cbG);
    lif_tile(c11, bmG + wm + 16, cbG + 16);
}

// ---------------- H = hsum / L ----------------
__global__ __launch_bounds__(256) void finalize_kernel(const float* __restrict__ hsum,
                                                       float* __restrict__ H) {
    size_t i = (size_t)blockIdx.x * 256 + threadIdx.x;
    H[i] = hsum[i] * INV_L_;
}

extern "C" void kernel_launch(void* const* d_in, const int* in_sizes, int n_in,
                              void* d_out, int out_size, void* d_ws, size_t ws_size,
                              hipStream_t stream) {
    const float* X = (const float*)d_in[0];   // [N, L, D]
    const float* W = (const float*)d_in[1];   // [D, T]
    const float* U = (const float*)d_in[2];   // [T, T]
    float* H = (float*)d_out;                 // [N, T]

    char* ws = (char*)d_ws;
    size_t off = 0;
    _Float16* X16 = (_Float16*)(ws + off); off += (size_t)N_ * L_ * D_ * 2;  // 210 MB
    _Float16* WT  = (_Float16*)(ws + off); off += (size_t)D_ * T_ * 2;       // 1 MB
    _Float16* UT  = (_Float16*)(ws + off); off += (size_t)T_ * T_ * 2;       // 2 MB
    float*    syn = (float*)(ws + off);    off += (size_t)N_ * T_ * 4;       // 8 MB
    float*    mem = (float*)(ws + off);    off += (size_t)N_ * T_ * 4;       // 8 MB
    float*    hsm = (float*)(ws + off);    off += (size_t)N_ * T_ * 4;       // 8 MB
    _Float16* spkA = (_Float16*)(ws + off); off += (size_t)N_ * T_ * 2;      // 4 MB
    _Float16* spkB = (_Float16*)(ws + off); off += (size_t)N_ * T_ * 2;      // 4 MB

    // Precision conversion / transposes (so B fragments are contiguous-in-K loads)
    cvt_x_kernel<<<(unsigned)((size_t)N_ * L_ * D_ / (256 * 4)), 256, 0, stream>>>(X, X16);
    transpose_cvt_kernel<<<(D_ * T_) / 256, 256, 0, stream>>>(W, WT, D_, T_);
    transpose_cvt_kernel<<<(T_ * T_) / 256, 256, 0, stream>>>(U, UT, T_, T_);

    // Zero syn, mem, hsum, spkA (contiguous in ws)
    {
        size_t zbytes = (size_t)N_ * T_ * (4 + 4 + 4 + 2);
        zero_kernel<<<(unsigned)(zbytes / 16 / 256), 256, 0, stream>>>((v4u*)syn);
    }

    // 100 sequential LIF steps (grid-wide dependency -> one launch per step)
    dim3 grid(N_ / 64, T_ / 128);
    for (int t = 0; t < L_; ++t) {
        const _Float16* cur = (t & 1) ? spkB : spkA;
        _Float16*       nxt = (t & 1) ? spkA : spkB;
        lif_step_kernel<<<grid, 256, 0, stream>>>(X16, WT, UT, cur, nxt, syn, mem, hsm, t);
    }

    finalize_kernel<<<(N_ * T_) / 256, 256, 0, stream>>>(hsm, H);
}
